// HybridEncoder_7043746365910
// MI455X (gfx1250) — matrix operand order repaired and checked by reading, not compiled, not run on hardware
//
#include <hip/hip_runtime.h>
#include <stdint.h>
#include <math.h>

// ---------------- types for WMMA ----------------
typedef __attribute__((ext_vector_type(16))) __bf16 v16bf;
typedef __attribute__((ext_vector_type(8)))  float  v8f;
typedef __attribute__((ext_vector_type(8)))  unsigned int v8u;
typedef int v4i_gcc __attribute__((vector_size(16)));   // matches builtin param type

__device__ __forceinline__ v16bf as_bf16(v8u u) {
    union { v8u u; v16bf b; } c; c.u = u; return c.b;
}
__device__ __forceinline__ v8f wmma_bf16(v8u a, v8u b, v8f c) {
    return __builtin_amdgcn_wmma_f32_16x16x32_bf16(
        false, as_bf16(a), false, as_bf16(b), (short)0, c, false, false);
}
__device__ __forceinline__ v8f zero8() {
    v8f z;
#pragma unroll
    for (int i = 0; i < 8; ++i) z[i] = 0.f;
    return z;
}
__device__ __forceinline__ unsigned short f2bf(float f) {
    unsigned int u = __builtin_bit_cast(unsigned int, f);
    unsigned int r = u + 0x7FFFu + ((u >> 16) & 1u);   // RNE
    return (unsigned short)(r >> 16);
}
__device__ __forceinline__ float gelu_f(float x) {
    return 0.5f * x * (1.0f + erff(x * 0.70710678118654752440f));
}

// ---------------- gfx1250 async global->LDS copy (guarded) ----------------
#if defined(__gfx1250__) && __has_builtin(__builtin_amdgcn_global_load_async_to_lds_b128)
#define HAS_ASYNC_LDS 1
#else
#define HAS_ASYNC_LDS 0
#endif

__device__ __forceinline__ void cp16_g2l(unsigned short* dst_lds, const unsigned short* src_g) {
#if HAS_ASYNC_LDS
    __builtin_amdgcn_global_load_async_to_lds_b128(
        (__attribute__((address_space(1))) v4i_gcc*)(void*)const_cast<unsigned short*>(src_g),
        (__attribute__((address_space(3))) v4i_gcc*)(void*)dst_lds,
        0, 0);
#else
    *(uint4*)dst_lds = *(const uint4*)src_g;
#endif
}
__device__ __forceinline__ void wait_async0() {
#if HAS_ASYNC_LDS
#if __has_builtin(__builtin_amdgcn_s_wait_asynccnt)
    __builtin_amdgcn_s_wait_asynccnt(0);
#else
    asm volatile("s_wait_asynccnt 0x0" ::: "memory");
#endif
#endif
}

// ---------------- constants ----------------
#define Bc   8
#define Sc   512
#define Dc   768
#define Lc   6
#define NFc  256
#define FFc  3072
#define Hc   12
#define Mtot 4096           // B*S
#define BM   128
#define BN   128
#define BK   32
#define LDA_S 40            // LDS A row stride (elements)
#define WT_S  34            // LDS transposed-W row stride (elements)

// ---------------- generic bf16 WMMA GEMM (double-buffered) ----------------
struct GemmP {
    const unsigned short* A;     // bf16 [rows, lda]
    const unsigned short* W;     // bf16 [K, N] row-major (ldw = N)
    const float* bias;           // [N] or null
    const float* Cin;            // accumulate source (ldc/colOff layout) or null
    const float* resid;          // residual add [M, ldr] or null
    float* Cf;                   // fp32 out or null
    unsigned short* Cb;          // bf16 out or null
    int N, K, lda, ldw;
    int seqLen, padExtra, rowOff; // A row map: (r/seq)*(seq+padExtra)+(r%seq)+rowOff
    int ldc, colOff, ldcb, colOffb, ldr, act;
};

__global__ __launch_bounds__(256) void gemm_bf16(GemmP p) {
    __shared__ __align__(16) unsigned short aS[2][BM * LDA_S];
    __shared__ __align__(16) unsigned short wS[2][BN * WT_S];
    const int tid  = threadIdx.x;
    const int wave = tid >> 5, lane = tid & 31;
    const int half = lane >> 4, hl = lane & 15;
    const int waveM = wave >> 1, waveN = wave & 1;  // 4 x 2 wave grid
    const int bm = blockIdx.x * BM, bn = blockIdx.y * BN;

    v8f acc[2][4];
#pragma unroll
    for (int mt = 0; mt < 2; ++mt)
#pragma unroll
        for (int t = 0; t < 4; ++t) acc[mt][t] = zero8();

    // stage A tile [128 rows x 32 k] into aS[buf] (async if available)
    auto stageA = [&](int k0, int buf) {
#pragma unroll
        for (int j = 0; j < 2; ++j) {
            int c = tid + 256 * j;          // 512 chunks of 8 elements
            int row = c >> 2, sub = c & 3;
            int r = bm + row;
            int arow = (r / p.seqLen) * (p.seqLen + p.padExtra) + (r % p.seqLen) + p.rowOff;
            const unsigned short* src = p.A + (size_t)arow * p.lda + k0 + sub * 8;
            cp16_g2l(&aS[buf][row * LDA_S + sub * 8], src);
        }
    };
    // stage W tile [32 k x 128 n] transposed into wS[buf][n][k]
    auto stageW = [&](int k0, int buf) {
#pragma unroll
        for (int j = 0; j < 2; ++j) {
            int c = tid + 256 * j;          // 512 chunks of 8 elements
            int kr = c >> 4, sub = c & 15;
            const unsigned short* src = p.W + (size_t)(k0 + kr) * p.ldw + bn + sub * 8;
            union { uint4 v; unsigned short s[8]; } t16;
            t16.v = *(const uint4*)src;
#pragma unroll
            for (int i = 0; i < 8; ++i)
                wS[buf][(sub * 8 + i) * WT_S + kr] = t16.s[i];
        }
    };

    const int nk = p.K / BK;
    stageA(0, 0); stageW(0, 0);
    wait_async0();
    __syncthreads();

    for (int ki = 0; ki < nk; ++ki) {
        const int buf = ki & 1;
        if (ki + 1 < nk) { stageA((ki + 1) * BK, buf ^ 1); stageW((ki + 1) * BK, buf ^ 1); }

        // fragments + 8 WMMAs (wave owns 32 rows x 64 cols)
        v8u af[2];
#pragma unroll
        for (int mt = 0; mt < 2; ++mt)
#pragma unroll
            for (int v = 0; v < 8; ++v) {
                int kb = 16 * half + 2 * v;
                af[mt][v] = *(const unsigned int*)&aS[buf][(waveM * 32 + mt * 16 + hl) * LDA_S + kb];
            }
#pragma unroll
        for (int t = 0; t < 4; ++t) {
            v8u bfr;
#pragma unroll
            for (int v = 0; v < 8; ++v) {
                int kb = 16 * half + 2 * v;
                bfr[v] = *(const unsigned int*)&wS[buf][(waveN * 64 + t * 16 + hl) * WT_S + kb];
            }
#pragma unroll
            for (int mt = 0; mt < 2; ++mt)
                acc[mt][t] = wmma_bf16(af[mt], bfr, acc[mt][t]);
        }
        wait_async0();
        __syncthreads();
    }

    // ---- epilogue ----
#pragma unroll
    for (int mt = 0; mt < 2; ++mt)
#pragma unroll
        for (int t = 0; t < 4; ++t) {
#pragma unroll
            for (int r = 0; r < 8; ++r) {
                int row = bm + waveM * 32 + mt * 16 + r + 8 * half;
                int col = bn + waveN * 64 + t * 16 + hl;
                float s = acc[mt][t][r];
                if (p.Cin)  s += p.Cin[(size_t)row * p.ldc + p.colOff + col];
                if (p.bias) s += p.bias[col];
                if (p.act)  s = gelu_f(s);
                if (p.resid) s += p.resid[(size_t)row * p.ldr + col];
                if (p.Cf) p.Cf[(size_t)row * p.ldc + p.colOff + col] = s;
                if (p.Cb) p.Cb[(size_t)row * p.ldcb + p.colOffb + col] = f2bf(s);
            }
        }
}

// ---------------- fused attention: one wave per (b, h, 16-query tile) ----------------
__global__ __launch_bounds__(32) void attn_kernel(
    const unsigned short* __restrict__ Q, const unsigned short* __restrict__ Kb,
    const unsigned short* __restrict__ vT, const int* __restrict__ mask,
    unsigned short* __restrict__ Ob, float scale)
{
    __shared__ __align__(16) float sc[16 * 520];
    __shared__ __align__(16) unsigned short pm[16 * 528];
    const int lane = threadIdx.x;
    const int half = lane >> 4, hl = lane & 15;
    const int qt = blockIdx.x, h = blockIdx.y, b = blockIdx.z;
    const int q0 = qt * 16;
    const size_t qkbase = ((size_t)b * Sc) * Dc + h * 64;

    // phase 1: scores = Q K^T * scale + mask_bias   (WMMA, dh=64 in 2 chunks)
    for (int k0 = 0; k0 < Sc; k0 += 16) {
        v8f acc = zero8();
#pragma unroll
        for (int c = 0; c < 2; ++c) {
            v8u af, bfr;
#pragma unroll
            for (int v = 0; v < 8; ++v) {
                int kb = 16 * half + 2 * v + 32 * c;
                af[v] = *(const unsigned int*)(Q + qkbase + (size_t)(q0 + hl) * Dc + kb);
                int kb2 = 16 * half + 2 * v + 32 * c;
                bfr[v] = *(const unsigned int*)(Kb + qkbase + (size_t)(k0 + hl) * Dc + kb2);
            }
            acc = wmma_bf16(af, bfr, acc);
        }
        int key = k0 + hl;
        float mb = (mask[b * Sc + key] == 0) ? -1e9f : 0.f;
#pragma unroll
        for (int r = 0; r < 8; ++r) {
            int q = r + 8 * half;
            sc[q * 520 + key] = acc[r] * scale + mb;
        }
    }
    __syncthreads();

    // phase 2: row softmax -> bf16 probabilities in LDS
    if (lane < 16) {
        const int q = lane;
        float m = -3.4e38f;
        for (int k = 0; k < Sc; ++k) m = fmaxf(m, sc[q * 520 + k]);
        float sum = 0.f;
        for (int k = 0; k < Sc; ++k) { float e = expf(sc[q * 520 + k] - m); sum += e; sc[q * 520 + k] = e; }
        float inv = 1.f / sum;
        for (int k = 0; k < Sc; ++k) pm[q * 528 + k] = f2bf(sc[q * 520 + k] * inv);
    }
    __syncthreads();

    // phase 3: O = P @ V  (V pre-transposed: vT[b][h][dh][s])
    v8f o[4];
#pragma unroll
    for (int t = 0; t < 4; ++t) o[t] = zero8();
    for (int k0 = 0; k0 < Sc; k0 += 32) {
        v8u af;
#pragma unroll
        for (int v = 0; v < 8; ++v) {
            int kb = 16 * half + 2 * v;
            af[v] = *(const unsigned int*)&pm[hl * 528 + k0 + kb];
        }
#pragma unroll
        for (int t = 0; t < 4; ++t) {
            v8u bfr;
#pragma unroll
            for (int v = 0; v < 8; ++v) {
                int kk = k0 + 16 * half + 2 * v;
                bfr[v] = *(const unsigned int*)(vT + (((size_t)b * Hc + h) * 64 + t * 16 + hl) * Sc + kk);
            }
            o[t] = wmma_bf16(af, bfr, o[t]);
        }
    }
#pragma unroll
    for (int t = 0; t < 4; ++t)
#pragma unroll
        for (int r = 0; r < 8; ++r) {
            int row = b * Sc + q0 + r + 8 * half;
            int col = h * 64 + t * 16 + hl;
            Ob[(size_t)row * Dc + col] = f2bf(o[t][r]);
        }
}

// ---------------- elementwise kernels ----------------
__global__ __launch_bounds__(256) void ln_kernel(const float* __restrict__ X,
    const float* __restrict__ g, const float* __restrict__ bta,
    float* outF, unsigned short* outB, int D)
{
    __shared__ float red[256];
    const int row = blockIdx.x, tid = threadIdx.x;
    const float* x = X + (size_t)row * D;
    float s = 0.f;
    for (int i = tid; i < D; i += 256) s += x[i];
    red[tid] = s; __syncthreads();
    for (int o = 128; o > 0; o >>= 1) { if (tid < o) red[tid] += red[tid + o]; __syncthreads(); }
    float mean = red[0] / D;
    __syncthreads();
    float v = 0.f;
    for (int i = tid; i < D; i += 256) { float d0 = x[i] - mean; v += d0 * d0; }
    red[tid] = v; __syncthreads();
    for (int o = 128; o > 0; o >>= 1) { if (tid < o) red[tid] += red[tid + o]; __syncthreads(); }
    float rstd = rsqrtf(red[0] / D + 1e-5f);
    for (int i = tid; i < D; i += 256) {
        float y = (x[i] - mean) * rstd * g[i] + bta[i];
        size_t idx = (size_t)row * D + i;
        if (outF) outF[idx] = y;
        if (outB) outB[idx] = f2bf(y);
    }
}

__global__ void embed_kernel(const int* __restrict__ ids, const float* __restrict__ emb,
                             const float* __restrict__ pos, float* xf, unsigned short* xb)
{
    int i = blockIdx.x * 256 + threadIdx.x;
    if (i >= Mtot * Dc) return;
    int row = i / Dc, d = i % Dc;
    int s = row & (Sc - 1);
    int tok = ids[row];
    float val = emb[(size_t)tok * Dc + d] + pos[(size_t)s * Dc + d];
    xf[i] = val; xb[i] = f2bf(val);
}

__global__ void xpad_kernel(const unsigned short* __restrict__ xb, unsigned short* xp)
{
    int i = blockIdx.x * 256 + threadIdx.x;
    const int total = Bc * (Sc + 6) * Dc;
    if (i >= total) return;
    int d = i % Dc, rs = i / Dc;
    int b = rs / (Sc + 6), sp = rs % (Sc + 6);
    unsigned short v = 0;
    if (sp >= 3 && sp < Sc + 3) v = xb[((size_t)b * Sc + (sp - 3)) * Dc + d];
    xp[i] = v;
}

__global__ void bn_gelu_kernel(const float* __restrict__ in, const float* __restrict__ g,
                               const float* __restrict__ bta, unsigned short* out)
{
    int i = blockIdx.x * 256 + threadIdx.x;
    if (i >= Mtot * Dc) return;
    int c = i % Dc;
    const float bn_rsd = rsqrtf(1.0f + 1e-5f);
    float v = in[i] * bn_rsd * g[c] + bta[c];
    out[i] = f2bf(gelu_f(v));
}

__global__ void transpose_v_kernel(const unsigned short* __restrict__ v, unsigned short* vT)
{
    int i = blockIdx.x * 256 + threadIdx.x;
    if (i >= Mtot * Dc) return;
    int row = i / Dc, col = i % Dc;
    int b = row >> 9, s = row & (Sc - 1);
    int h = col >> 6, dh = col & 63;
    vT[(((size_t)b * Hc + h) * 64 + dh) * Sc + s] = v[i];
}

__global__ void cvt_kernel(const float* __restrict__ in, unsigned short* out, int n)
{
    int i = blockIdx.x * 256 + threadIdx.x;
    if (i < n) out[i] = f2bf(in[i]);
}

// conv weights: in [NF, D, Kk] (one layer) -> out [Kk][D][NF] bf16
__global__ void convw_kernel(const float* __restrict__ in, unsigned short* out, int Kk)
{
    int i = blockIdx.x * 256 + threadIdx.x;
    int total = Kk * Dc * NFc;
    if (i >= total) return;
    int nf = i % NFc, rem = i / NFc;
    int d = rem % Dc, k = rem / Dc;
    out[i] = f2bf(in[((size_t)nf * Dc + d) * Kk + k]);
}

// ---------------- host orchestration ----------------
static inline int cdiv(int a, int b) { return (a + b - 1) / b; }

static void launch_gemm(hipStream_t st, const unsigned short* A, int lda, int seq, int padE, int rowOff,
                        const unsigned short* W, int K, int N, const float* bias, const float* Cin,
                        const float* resid, int ldr, float* Cf, int ldc, int colOff,
                        unsigned short* Cb, int ldcb, int colOffb, int act)
{
    GemmP p;
    p.A = A; p.W = W; p.bias = bias; p.Cin = Cin; p.resid = resid; p.Cf = Cf; p.Cb = Cb;
    p.N = N; p.K = K; p.lda = lda; p.ldw = N;
    p.seqLen = seq; p.padExtra = padE; p.rowOff = rowOff;
    p.ldc = ldc; p.colOff = colOff; p.ldcb = ldcb; p.colOffb = colOffb; p.ldr = ldr; p.act = act;
    dim3 grid(Mtot / BM, N / BN);
    gemm_bf16<<<grid, 256, 0, st>>>(p);
}

extern "C" void kernel_launch(void* const* d_in, const int* in_sizes, int n_in,
                              void* d_out, int out_size, void* d_ws, size_t ws_size,
                              hipStream_t stream)
{
    (void)in_sizes; (void)n_in; (void)out_size; (void)ws_size;
    const int*   ids   = (const int*)d_in[0];
    const int*   amask = (const int*)d_in[1];
    const float* emb   = (const float*)d_in[2];
    const float* pos   = (const float*)d_in[3];
    const float* cw[3] = { (const float*)d_in[4], (const float*)d_in[6], (const float*)d_in[8] };
    const float* cb[3] = { (const float*)d_in[5], (const float*)d_in[7], (const float*)d_in[9] };
    const float* bng   = (const float*)d_in[10];
    const float* bnb   = (const float*)d_in[11];
    const float* cpw   = (const float*)d_in[12];
    const float* cpb   = (const float*)d_in[13];
    const float* ln1g  = (const float*)d_in[14];
    const float* ln1b  = (const float*)d_in[15];
    const float* ln2g  = (const float*)d_in[16];
    const float* ln2b  = (const float*)d_in[17];
    const float* wq    = (const float*)d_in[18];
    const float* bq    = (const float*)d_in[19];
    const float* wk    = (const float*)d_in[20];
    const float* bk    = (const float*)d_in[21];
    const float* wv    = (const float*)d_in[22];
    const float* bv    = (const float*)d_in[23];
    const float* wo    = (const float*)d_in[24];
    const float* bo    = (const float*)d_in[25];
    const float* fw1   = (const float*)d_in[26];
    const float* fb1   = (const float*)d_in[27];
    const float* fw2   = (const float*)d_in[28];
    const float* fb2   = (const float*)d_in[29];
    const float* fuw   = (const float*)d_in[30];
    const float* fub   = (const float*)d_in[31];
    const float* flg   = (const float*)d_in[32];
    const float* flb   = (const float*)d_in[33];
    const float* oww   = (const float*)d_in[34];
    const float* obb   = (const float*)d_in[35];
    const float* olg   = (const float*)d_in[36];
    const float* olb   = (const float*)d_in[37];

    // workspace carve-up
    size_t off = 0;
    char* base = (char*)d_ws;
    auto alloc = [&](size_t bytes) -> void* {
        off = (off + 255) & ~(size_t)255;
        void* p = base + off;
        off += bytes;
        return p;
    };
    const size_t MD = (size_t)Mtot * Dc;
    float*          xf      = (float*)alloc(MD * 4);
    unsigned short* xb      = (unsigned short*)alloc(MD * 2);
    unsigned short* xpad    = (unsigned short*)alloc((size_t)Bc * (Sc + 6) * Dc * 2);
    float*          convacc = (float*)alloc(MD * 4);              // also reused: fusion out, logits
    unsigned short* convact = (unsigned short*)alloc(MD * 2);
    unsigned short* convW   = (unsigned short*)alloc((size_t)15 * Dc * NFc * 2);
    unsigned short* wbuf    = (unsigned short*)alloc((size_t)FFc * Dc * 2);
    unsigned short* hb      = (unsigned short*)alloc(MD * 2);
    unsigned short* qb      = (unsigned short*)alloc(MD * 2);
    unsigned short* kb_     = (unsigned short*)alloc(MD * 2);
    unsigned short* vb      = (unsigned short*)alloc(MD * 2);
    unsigned short* vT      = (unsigned short*)alloc(MD * 2);
    unsigned short* aob     = (unsigned short*)alloc(MD * 2);
    float*          tf      = (float*)alloc(MD * 4);
    unsigned short* h2b     = (unsigned short*)alloc(MD * 2);
    unsigned short* ffb     = (unsigned short*)alloc((size_t)Mtot * FFc * 2);
    unsigned short* fusedA  = (unsigned short*)alloc((size_t)Mtot * 2 * Dc * 2);

    const int EW = 256;
    const int gMD = cdiv(Mtot * Dc, EW);

    // x = emb[ids] + pos
    embed_kernel<<<gMD, EW, 0, stream>>>(ids, emb, pos, xf, xb);

    const int KSs[3] = { 3, 5, 7 };
    unsigned short* convWk[3] = { convW, convW + (size_t)3 * Dc * NFc, convW + (size_t)8 * Dc * NFc };

    for (int l = 0; l < Lc; ++l) {
        // ---- conv branch ----
        xpad_kernel<<<cdiv(Bc * (Sc + 6) * Dc, EW), EW, 0, stream>>>(xb, xpad);
        for (int i = 0; i < 3; ++i) {
            int Kk = KSs[i];
            convw_kernel<<<cdiv(Kk * Dc * NFc, EW), EW, 0, stream>>>(
                cw[i] + (size_t)l * NFc * Dc * Kk, convWk[i], Kk);
        }
        for (int i = 0; i < 3; ++i) {
            int Kk = KSs[i];
            for (int tap = 0; tap < Kk; ++tap) {
                launch_gemm(stream, xpad, Dc, Sc, 6, tap + 3 - Kk / 2,
                            convWk[i] + (size_t)tap * Dc * NFc, Dc, NFc,
                            (tap == 0) ? (cb[i] + (size_t)l * NFc) : nullptr,
                            (tap == 0) ? nullptr : convacc,
                            nullptr, 0,
                            convacc, Dc, i * NFc,
                            nullptr, 0, 0, 0);
            }
        }
        bn_gelu_kernel<<<gMD, EW, 0, stream>>>(convacc, bng + (size_t)l * 3 * NFc,
                                               bnb + (size_t)l * 3 * NFc, convact);
        cvt_kernel<<<cdiv(Dc * Dc, EW), EW, 0, stream>>>(cpw + (size_t)l * Dc * Dc, wbuf, Dc * Dc);
        launch_gemm(stream, convact, Dc, Sc, 0, 0, wbuf, Dc, Dc,
                    cpb + (size_t)l * Dc, nullptr, xf, Dc,
                    nullptr, Dc, 0, fusedA, 2 * Dc, 0, 0);     // conv_out -> fused cols [0,768)

        // ---- transformer branch ----
        ln_kernel<<<Mtot, 256, 0, stream>>>(xf, ln1g + (size_t)l * Dc, ln1b + (size_t)l * Dc,
                                            nullptr, hb, Dc);
        cvt_kernel<<<cdiv(Dc * Dc, EW), EW, 0, stream>>>(wq + (size_t)l * Dc * Dc, wbuf, Dc * Dc);
        launch_gemm(stream, hb, Dc, Sc, 0, 0, wbuf, Dc, Dc, bq + (size_t)l * Dc,
                    nullptr, nullptr, 0, nullptr, Dc, 0, qb, Dc, 0, 0);
        cvt_kernel<<<cdiv(Dc * Dc, EW), EW, 0, stream>>>(wk + (size_t)l * Dc * Dc, wbuf, Dc * Dc);
        launch_gemm(stream, hb, Dc, Sc, 0, 0, wbuf, Dc, Dc, bk + (size_t)l * Dc,
                    nullptr, nullptr, 0, nullptr, Dc, 0, kb_, Dc, 0, 0);
        cvt_kernel<<<cdiv(Dc * Dc, EW), EW, 0, stream>>>(wv + (size_t)l * Dc * Dc, wbuf, Dc * Dc);
        launch_gemm(stream, hb, Dc, Sc, 0, 0, wbuf, Dc, Dc, bv + (size_t)l * Dc,
                    nullptr, nullptr, 0, nullptr, Dc, 0, vb, Dc, 0, 0);

        transpose_v_kernel<<<gMD, EW, 0, stream>>>(vb, vT);
        attn_kernel<<<dim3(Sc / 16, Hc, Bc), 32, 0, stream>>>(qb, kb_, vT, amask, aob, 0.125f);

        cvt_kernel<<<cdiv(Dc * Dc, EW), EW, 0, stream>>>(wo + (size_t)l * Dc * Dc, wbuf, Dc * Dc);
        launch_gemm(stream, aob, Dc, Sc, 0, 0, wbuf, Dc, Dc, bo + (size_t)l * Dc,
                    nullptr, xf, Dc, tf, Dc, 0, nullptr, 0, 0, 0);   // t = x + ao

        ln_kernel<<<Mtot, 256, 0, stream>>>(tf, ln2g + (size_t)l * Dc, ln2b + (size_t)l * Dc,
                                            nullptr, h2b, Dc);
        cvt_kernel<<<cdiv(Dc * FFc, EW), EW, 0, stream>>>(fw1 + (size_t)l * Dc * FFc, wbuf, Dc * FFc);
        launch_gemm(stream, h2b, Dc, Sc, 0, 0, wbuf, Dc, FFc, fb1 + (size_t)l * FFc,
                    nullptr, nullptr, 0, nullptr, FFc, 0, ffb, FFc, 0, 1); // GELU
        cvt_kernel<<<cdiv(FFc * Dc, EW), EW, 0, stream>>>(fw2 + (size_t)l * FFc * Dc, wbuf, FFc * Dc);
        launch_gemm(stream, ffb, FFc, Sc, 0, 0, wbuf, FFc, Dc, fb2 + (size_t)l * Dc,
                    nullptr, tf, Dc, nullptr, Dc, 0, fusedA, 2 * Dc, Dc, 0); // trans_out -> fused cols [768,1536)

        // ---- fusion ----
        cvt_kernel<<<cdiv(2 * Dc * Dc, EW), EW, 0, stream>>>(fuw + (size_t)l * 2 * Dc * Dc, wbuf, 2 * Dc * Dc);
        launch_gemm(stream, fusedA, 2 * Dc, Sc, 0, 0, wbuf, 2 * Dc, Dc, fub + (size_t)l * Dc,
                    nullptr, nullptr, 0, convacc, Dc, 0, nullptr, 0, 0, 0);
        ln_kernel<<<Mtot, 256, 0, stream>>>(convacc, flg + (size_t)l * Dc, flb + (size_t)l * Dc,
                                            xf, xb, Dc);
    }

    // output head: LN(x @ ow + ob)
    cvt_kernel<<<cdiv(Dc * Dc, EW), EW, 0, stream>>>(oww, wbuf, Dc * Dc);
    launch_gemm(stream, xb, Dc, Sc, 0, 0, wbuf, Dc, Dc, obb,
                nullptr, nullptr, 0, convacc, Dc, 0, nullptr, 0, 0, 0);
    ln_kernel<<<Mtot, 256, 0, stream>>>(convacc, olg, olb, (float*)d_out, nullptr, Dc);
}
